// Attention_67714454389028
// MI455X (gfx1250) — compile-verified, matrix operand order checked by reading, not run on hardware
//
#include <hip/hip_runtime.h>
#include <hip/hip_bf16.h>

// Problem constants (from reference): B=32, N=4096, C=768, H=8, DH=96
#define BB 32
#define NN 4096
#define CC 768
#define HH 8
#define DHH 96
#define HPAD 16            // heads padded to 16 for the WMMA N-dim
#define ROWS 128           // x rows per block (chunk)
#define CHUNKS (NN / ROWS) // 32 chunks per batch

typedef __attribute__((ext_vector_type(16))) __bf16 v16bf;
typedef __attribute__((ext_vector_type(8)))  __bf16 v8bf;
typedef __attribute__((ext_vector_type(8)))  float  v8f;

// ---------------------------------------------------------------------------
// K1: fold query through Wq and Wk:
//   q[f]    = sum_c queries[c] * Wq[f, c]
//   r[h, c] = sum_d q[h*DH+d] * Wkv[h*DH+d, c]      (bf16, rows h in [8,16) = 0)
// One block, 256 threads. Tiny.
// ---------------------------------------------------------------------------
__global__ void fold_q_kernel(const float* __restrict__ queries,
                              const float* __restrict__ Wq,
                              const float* __restrict__ Wkv,
                              __bf16* __restrict__ rbf) {
    __shared__ float q[CC];
    const int t = threadIdx.x;
    for (int f = t; f < CC; f += 256) {
        const float* wr = Wq + (size_t)f * CC;
        float acc = 0.f;
        for (int c = 0; c < CC; ++c) acc += queries[c] * wr[c];
        q[f] = acc;
    }
    __syncthreads();
    for (int idx = t; idx < HPAD * CC; idx += 256) {
        const int h = idx / CC;
        const int c = idx - h * CC;
        float acc = 0.f;
        if (h < HH) {
            const float* w  = Wkv + (size_t)(h * DHH) * CC + c;
            const float* qq = q + h * DHH;
            for (int d = 0; d < DHH; ++d) acc += qq[d] * w[(size_t)d * CC];
        }
        rbf[idx] = (__bf16)acc;   // rows 8..15 zero -> WMMA pad columns produce 0
    }
}

// ---------------------------------------------------------------------------
// K2 (fused, flash-style): one block per (b, 128-row chunk).
//  P1: dots[n,h] = r[h,:].x[b,n,:] via v_wmma_f32_16x16x32_bf16 -> LDS
//  P2: per-head LOCAL softmax numerators (wave-level shfl reductions;
//      wave w owns head h=w), keep local max m_h and local sum l_h
//  P3: s_loc[h][c] = sum_n exp(d-m_h) * x[b,n,c]  (x re-read is L2-hot)
//  Writes unnormalized partials (s_loc, m, l) for the merge kernel.
// ---------------------------------------------------------------------------
__global__ void fused_attn_kernel(const float* __restrict__ x,
                                  const __bf16* __restrict__ rbf,
                                  float* __restrict__ part_s,
                                  float* __restrict__ part_m,
                                  float* __restrict__ part_l) {
    const int b     = blockIdx.x / CHUNKS;
    const int chunk = blockIdx.x - b * CHUNKS;
    const int n0    = chunk * ROWS;

    const int wv   = threadIdx.x >> 5;
    const int ln   = threadIdx.x & 31;
    const int row  = ln & 15;            // M index (A) / N index (B)
    const int hi   = ln >> 4;            // K half selector
    const int coff = hi * 8;

    __shared__ float lds_w[HH][ROWS];    // dots, then exp-weights
    __shared__ float m_sh[HH], l_sh[HH];

    // ---- Phase 1: WMMA dot tiles (16 rows per wave) ----
    const float*  xrow = x   + ((size_t)b * NN + (n0 + wv * 16 + row)) * CC;
    const __bf16* rrow = rbf + (size_t)row * CC;

    v8f dacc = {};
    for (int c0 = 0; c0 < CC; c0 += 32) {
        const float* xa = xrow + c0 + coff;
        v16bf a;
#pragma unroll
        for (int e = 0; e < 8; ++e) {
            // 16-bit A layout: lanes 0-15: K=0..7 (v0..3), 16..23 (v4..7)
            //                  lanes 16-31: K=8..15,       24..31
            a[e]     = (__bf16)xa[e];
            a[8 + e] = (__bf16)xa[16 + e];
        }
        v8bf blo = *(const v8bf*)(rrow + c0 + coff);
        v8bf bhi = *(const v8bf*)(rrow + c0 + 16 + coff);
        v16bf bm = __builtin_shufflevector(blo, bhi,
                       0, 1, 2, 3, 4, 5, 6, 7, 8, 9, 10, 11, 12, 13, 14, 15);
        dacc = __builtin_amdgcn_wmma_f32_16x16x32_bf16(
            false, a, false, bm, (short)0, dacc, false, false);
    }

    // C/D layout: lane<16 -> (N=lane, M=vgpr); lane>=16 -> (N=lane-16, M=8+vgpr)
    const int h8    = ln & 15;
    const int mbase = hi * 8;
    if (h8 < HH) {
#pragma unroll
        for (int v = 0; v < 8; ++v)
            lds_w[h8][wv * 16 + mbase + v] = dacc[v];
    }
    __syncthreads();

    // ---- Phase 2: per-head local softmax numerators (wave wv -> head wv) ----
    {
        const int h = wv;                 // exactly 8 waves, 8 heads
        float v0 = lds_w[h][ln * 4 + 0];
        float v1 = lds_w[h][ln * 4 + 1];
        float v2 = lds_w[h][ln * 4 + 2];
        float v3 = lds_w[h][ln * 4 + 3];
        float m = fmaxf(fmaxf(v0, v1), fmaxf(v2, v3));
#pragma unroll
        for (int off = 16; off > 0; off >>= 1)
            m = fmaxf(m, __shfl_xor(m, off, 32));
        float e0 = __expf(v0 - m), e1 = __expf(v1 - m);
        float e2 = __expf(v2 - m), e3 = __expf(v3 - m);
        float sum = (e0 + e1) + (e2 + e3);
#pragma unroll
        for (int off = 16; off > 0; off >>= 1)
            sum += __shfl_xor(sum, off, 32);
        lds_w[h][ln * 4 + 0] = e0;
        lds_w[h][ln * 4 + 1] = e1;
        lds_w[h][ln * 4 + 2] = e2;
        lds_w[h][ln * 4 + 3] = e3;
        if (ln == 0) { m_sh[h] = m; l_sh[h] = sum; }
    }
    __syncthreads();

    // ---- Phase 3: weighted accumulation over the (L2-hot) chunk ----
    const int t = threadIdx.x;            // column base; owns c = t, t+256, t+512
    float acc[3][HH];
#pragma unroll
    for (int j = 0; j < 3; ++j)
#pragma unroll
        for (int h = 0; h < HH; ++h) acc[j][h] = 0.f;

    const float* xb = x + ((size_t)b * NN + n0) * CC;
    for (int nn = 0; nn < ROWS; ++nn) {
        float wvh[HH];
#pragma unroll
        for (int h = 0; h < HH; ++h) wvh[h] = lds_w[h][nn];   // LDS broadcast
        const float* xr = xb + (size_t)nn * CC;
#pragma unroll
        for (int j = 0; j < 3; ++j) {
            const float xv = xr[t + j * 256];
#pragma unroll
            for (int h = 0; h < HH; ++h) acc[j][h] += wvh[h] * xv;
        }
    }

    float* ps = part_s + (size_t)(b * CHUNKS + chunk) * HH * CC;
#pragma unroll
    for (int h = 0; h < HH; ++h)
#pragma unroll
        for (int j = 0; j < 3; ++j)
            ps[(size_t)h * CC + t + j * 256] = acc[j][h];
    if (t < HH) {
        part_m[(b * CHUNKS + chunk) * HH + t] = m_sh[t];
        part_l[(b * CHUNKS + chunk) * HH + t] = l_sh[t];
    }
}

// ---------------------------------------------------------------------------
// K3: merge chunk partials with log-sum-exp rescale:
//   m_g = max_k m_k ;  L = sum_k l_k*exp(m_k-m_g)
//   s[b,h,c] = sum_k exp(m_k-m_g)*s_k[h,c] / L
// ---------------------------------------------------------------------------
__global__ void merge_kernel(const float* __restrict__ part_s,
                             const float* __restrict__ part_m,
                             const float* __restrict__ part_l,
                             float* __restrict__ s) {
    const int b = blockIdx.x;
    const int t = threadIdx.x;
    __shared__ float msh[HH][CHUNKS], lsh[HH][CHUNKS];
    __shared__ float scale[CHUNKS][HH];
    __shared__ float invL[HH];

    {   // 256 threads == CHUNKS*HH
        const int k = t >> 3, h = t & 7;
        msh[h][k] = part_m[(b * CHUNKS + k) * HH + h];
        lsh[h][k] = part_l[(b * CHUNKS + k) * HH + h];
    }
    __syncthreads();
    if (t < HH) {
        float mg = -INFINITY;
        for (int k = 0; k < CHUNKS; ++k) mg = fmaxf(mg, msh[t][k]);
        float L = 0.f;
        for (int k = 0; k < CHUNKS; ++k) {
            const float sc = __expf(msh[t][k] - mg);
            scale[k][t] = sc;
            L += lsh[t][k] * sc;
        }
        invL[t] = 1.0f / L;
    }
    __syncthreads();

    for (int c = t; c < CC; c += 256) {
#pragma unroll
        for (int h = 0; h < HH; ++h) {
            float acc = 0.f;
            for (int k = 0; k < CHUNKS; ++k)
                acc += part_s[((size_t)(b * CHUNKS + k) * HH + h) * CC + c] *
                       scale[k][h];
            s[((size_t)b * HH + h) * CC + c] = acc * invL[h];
        }
    }
}

// ---------------------------------------------------------------------------
// K4: o[b, f] = Wv[f,:] . s[b, f/DH, :]   where Wv = Wkv rows [C, 2C)
// ---------------------------------------------------------------------------
__global__ void vproj_kernel(const float* __restrict__ Wkv,
                             const float* __restrict__ s,
                             float* __restrict__ o) {
    const int b = blockIdx.x;
    for (int f = threadIdx.x; f < CC; f += 256) {
        const int h = f / DHH;
        const float* wr = Wkv + (size_t)(CC + f) * CC;
        const float* sr = s + ((size_t)b * HH + h) * CC;
        float acc = 0.f;
        for (int c = 0; c < CC; ++c) acc += wr[c] * sr[c];
        o[(size_t)b * CC + f] = acc;
    }
}

// K5: out[b, e] = bproj[e] + Wproj[e,:] . o[b,:]
__global__ void proj_kernel(const float* __restrict__ Wproj,
                            const float* __restrict__ bproj,
                            const float* __restrict__ o,
                            float* __restrict__ out) {
    const int b = blockIdx.x;
    const float* ob = o + (size_t)b * CC;
    for (int e = threadIdx.x; e < CC; e += 256) {
        const float* wr = Wproj + (size_t)e * CC;
        float acc = bproj[e];
        for (int c = 0; c < CC; ++c) acc += wr[c] * ob[c];
        out[(size_t)b * CC + e] = acc;
    }
}

// ---------------------------------------------------------------------------
extern "C" void kernel_launch(void* const* d_in, const int* in_sizes, int n_in,
                              void* d_out, int out_size, void* d_ws, size_t ws_size,
                              hipStream_t stream) {
    const float* x       = (const float*)d_in[0];
    const float* queries = (const float*)d_in[1];
    const float* Wq      = (const float*)d_in[2];
    const float* Wkv     = (const float*)d_in[3];
    const float* Wproj   = (const float*)d_in[4];
    const float* bproj   = (const float*)d_in[5];
    float* out = (float*)d_out;

    // workspace carve-up (float units): rbf | part_s | part_m | part_l | s | o
    float*  ws     = (float*)d_ws;
    __bf16* rbf    = (__bf16*)ws;                          // HPAD*CC bf16 (24 KB)
    float*  part_s = ws + 8192;                            // B*CHUNKS*H*C (25 MB)
    float*  part_m = part_s + (size_t)BB * CHUNKS * HH * CC;
    float*  part_l = part_m + (size_t)BB * CHUNKS * HH;
    float*  s      = part_l + (size_t)BB * CHUNKS * HH;    // B*H*C
    float*  o      = s + (size_t)BB * HH * CC;             // B*C

    fold_q_kernel<<<1, 256, 0, stream>>>(queries, Wq, Wkv, rbf);

    fused_attn_kernel<<<dim3(BB * CHUNKS), 256, 0, stream>>>(
        x, rbf, part_s, part_m, part_l);

    merge_kernel<<<dim3(BB), 256, 0, stream>>>(part_s, part_m, part_l, s);

    vproj_kernel<<<dim3(BB), 256, 0, stream>>>(Wkv, s, o);

    proj_kernel<<<dim3(BB), 256, 0, stream>>>(Wproj, bproj, o, out);
}
